// CondConv2d_43447889166962
// MI455X (gfx1250) — matrix operand order repaired and checked by reading, not compiled, not run on hardware
//
#include <hip/hip_runtime.h>
#include <hip/hip_bf16.h>
#include <cstdint>

typedef __attribute__((ext_vector_type(16))) _Float16 v16h;
typedef __attribute__((ext_vector_type(8)))  _Float16 v8h;
typedef __attribute__((ext_vector_type(8)))  float    v8f;
typedef __attribute__((ext_vector_type(4)))  unsigned int uint4v;
typedef __attribute__((ext_vector_type(8)))  int int8v;
typedef __attribute__((ext_vector_type(4)))  int int4v;

union Frag { v16h v; v8h h[2]; };

static constexpr int C_IN   = 128;
static constexpr int C_OUT  = 256;
static constexpr int NEXP   = 8;
static constexpr int IMG_H  = 56;
static constexpr int IMG_W  = 56;
static constexpr int HWPIX  = IMG_H * IMG_W;          // 3136
static constexpr int NB     = 32;
static constexpr int KDIM   = 9 * C_IN;               // 1152, layout [p=kh*3+kw][i]
static constexpr int NPARAMS = C_OUT * KDIM;          // 294912 per sample
static constexpr int KTILES  = KDIM / 32;             // 36

// ---------------- workspace layout (float units) ----------------
// [0,     4096)  pooled  [B][C_IN]
// [4096,  4352)  routing [B][NEXP]
// [4352, 12544)  bias_b  [B][C_OUT]
// [12544, ... )  w16 (_Float16) [B][C_OUT][9][C_IN]  (18.9 MB)

// ---- 1) global average pool: one block per (c,b) -------------------------
__global__ __launch_bounds__(256) void cc_pool(const float* __restrict__ x,
                                               float* __restrict__ pooled) {
    const int c = blockIdx.x, b = blockIdx.y;
    const float* p = x + ((size_t)b * C_IN + c) * HWPIX;
    float s = 0.f;
    for (int i = threadIdx.x; i < HWPIX; i += 256) s += p[i];
    __shared__ float red[256];
    red[threadIdx.x] = s;
    __syncthreads();
    for (int st = 128; st > 0; st >>= 1) {
        if (threadIdx.x < st) red[threadIdx.x] += red[threadIdx.x + st];
        __syncthreads();
    }
    if (threadIdx.x == 0) pooled[b * C_IN + c] = red[0] * (1.0f / HWPIX);
}

// ---- 2) routing = sigmoid(pooled @ Wr^T + br); bias_b = routing @ bias ----
__global__ __launch_bounds__(256) void cc_route(const float* __restrict__ pooled,
                                                const float* __restrict__ rw,
                                                const float* __restrict__ rb,
                                                const float* __restrict__ bias,
                                                float* __restrict__ routing,
                                                float* __restrict__ bias_b) {
    const int b = blockIdx.x;
    __shared__ float ps[C_IN];
    __shared__ float rs[NEXP];
    if (threadIdx.x < C_IN) ps[threadIdx.x] = pooled[b * C_IN + threadIdx.x];
    __syncthreads();
    if (threadIdx.x < NEXP) {
        const int e = threadIdx.x;
        float z = rb[e];
        for (int c = 0; c < C_IN; ++c) z += ps[c] * rw[e * C_IN + c];
        const float r = 1.0f / (1.0f + __expf(-z));
        rs[e] = r;
        routing[b * NEXP + e] = r;
    }
    __syncthreads();
    const int o = threadIdx.x;
    if (o < C_OUT) {
        float s = 0.f;
#pragma unroll
        for (int e = 0; e < NEXP; ++e) s += rs[e] * bias[e * C_OUT + o];
        bias_b[b * C_OUT + o] = s;
    }
}

// ---- 3) mix expert weights -> f16, relayout [O][i][p] -> [O][p][i] -------
__global__ __launch_bounds__(256) void cc_mixw(const float* __restrict__ weight,
                                               const float* __restrict__ routing,
                                               _Float16* __restrict__ w16) {
    const int b = blockIdx.y;
    const int d = blockIdx.x * 256 + threadIdx.x;  // dest flat idx: o*1152 + p*128 + i
    __shared__ float rs[NEXP];
    if (threadIdx.x < NEXP) rs[threadIdx.x] = routing[b * NEXP + threadIdx.x];
    __syncthreads();
    const int o   = d / KDIM;
    const int rem = d - o * KDIM;
    const int p   = rem >> 7;
    const int i   = rem & 127;
    const int s   = o * KDIM + i * 9 + p;
    float acc = 0.f;
#pragma unroll
    for (int e = 0; e < NEXP; ++e) acc += rs[e] * weight[(size_t)e * NPARAMS + s];
    w16[(size_t)b * NPARAMS + d] = (_Float16)acc;
}

// ---- TDM: move one 256x32-half weight tile (row stride 1152 halves) to LDS,
//      with descriptor-generated padding: 16B pad every 64B -> 40-half rows.
__device__ __forceinline__ void tdm_load_wtile(const _Float16* gsrc, unsigned lds_off) {
    const unsigned long long ga = (unsigned long long)(uintptr_t)gsrc;
    uint4v g0;
    g0.x = 1u;                                   // count=1, user descriptor
    g0.y = lds_off;                              // [63:32] lds_addr (bytes)
    g0.z = (unsigned)ga;                         // [95:64] global_addr lo
    g0.w = (unsigned)(ga >> 32) | (2u << 30);    // [120:96] addr hi | type=2
    int8v g1;
    g1[0] = (1 << 16)        // data_size = 2B
          | (1 << 20)        // pad_enable
          | (3 << 22)        // pad_interval: 16 DWORDs = 64B
          | (3 << 25);       // pad_amount: 4 DWORDs = 16B
    g1[1] = (32 << 16);      // tensor_dim0 = 32 (bits 79:48, low half)
    g1[2] = (256 << 16);     // tensor_dim0 hi = 0 | tensor_dim1 = 256 (bits 111:80)
    g1[3] = (32 << 16);      // tensor_dim1 hi = 0 | tile_dim0 = 32 (bits 127:112)
    g1[4] = 256;             // tile_dim1 = 256 | tile_dim2 = 0
    g1[5] = (int)KDIM;       // tensor_dim0_stride = 1152 (bits 207:160, low)
    g1[6] = 0;
    g1[7] = 0;
    const int4v z4 = {0, 0, 0, 0};
    const int8v z8 = {0, 0, 0, 0, 0, 0, 0, 0};
    __builtin_amdgcn_tensor_load_to_lds(g0, g1, z4, z4, z8, 0);
}

// ---- 4) implicit-GEMM conv via v_wmma_f32_16x16x32_f16 --------------------
// Block tile: full M=256 x N=64; 8 waves, wave w owns M-blocks {w, w+8} x 4
// N-blocks -> 8 WMMAs per wave per K-step. A tiles: TDM double buffer in LDS;
// B tiles: coalesced im2col gather, register-staged.
__global__ __launch_bounds__(256) void cc_conv_wmma(const float* __restrict__ x,
                                                    const _Float16* __restrict__ w16,
                                                    const float* __restrict__ bias_b,
                                                    float* __restrict__ out) {
    __shared__ __align__(16) _Float16 As[2][256 * 40];   // 2 x 20 KB (TDM-filled)
    __shared__ __align__(16) _Float16 Bs[64 * 40];       // 5 KB

    const int tid  = threadIdx.x;
    const int b    = blockIdx.y;
    const int n0   = blockIdx.x * 64;
    const int lane = tid & 31;
    const int wave = tid >> 5;

    // B staging: pixel n = tid&63, 8 K-values at kk = (tid>>6)*8 (coalesced)
    const int bn  = tid & 63;
    const int bkk = (tid >> 6) * 8;
    const int hw  = n0 + bn;
    const int oh  = hw / IMG_W;
    const int ow  = hw - oh * IMG_W;

    const _Float16* wbase = w16 + (size_t)b * NPARAMS;
    const float*    xb    = x + (size_t)b * C_IN * HWPIX;

    unsigned as_lds[2];
    as_lds[0] = (unsigned)(uintptr_t)(__attribute__((address_space(3))) void*)&As[0][0];
    as_lds[1] = (unsigned)(uintptr_t)(__attribute__((address_space(3))) void*)&As[1][0];

    v8f acc[2][4] = {};
    v8h bReg;

    auto loadB = [&](int k0) {
        const int p  = k0 >> 7;
        const int i0 = k0 & 127;
        const int kh = p / 3, kw = p - kh * 3;
        const int ih = oh + kh - 1;
        const int iw = ow + kw - 1;
        const bool inb = ((unsigned)ih < (unsigned)IMG_H) & ((unsigned)iw < (unsigned)IMG_W);
        const float* src = xb + (size_t)(i0 + bkk) * HWPIX + ih * IMG_W + iw;
        v8h t;
#pragma unroll
        for (int j = 0; j < 8; ++j) {
            const float v = inb ? src[(size_t)j * HWPIX] : 0.0f;
            t[j] = (_Float16)v;
        }
        bReg = t;
    };

    // prologue: A tile 0 via TDM, B tile 0 into registers
    if (wave == 0) tdm_load_wtile(wbase, as_lds[0]);
    loadB(0);

    for (int kt = 0; kt < KTILES; ++kt) {
        const int buf = kt & 1;

        // publish B tile kt
        *(v8h*)&Bs[bn * 40 + bkk] = bReg;

        // TDM: start A tile kt+1 into the other buffer; wait for tile kt
        if (wave == 0) {
            if (kt + 1 < KTILES) {
                tdm_load_wtile(wbase + (kt + 1) * 32, as_lds[buf ^ 1]);
                __builtin_amdgcn_s_wait_tensorcnt(1);   // tile kt complete (in-order)
            } else {
                __builtin_amdgcn_s_wait_tensorcnt(0);
            }
        }
        __syncthreads();

        // stage next B tile under this step's WMMAs
        if (kt + 1 < KTILES) loadB((kt + 1) * 32);
        if (kt + 2 < KTILES) {
            const int k2  = (kt + 2) * 32;
            const int p2  = k2 >> 7;
            const int kh2 = p2 / 3, kw2 = p2 - kh2 * 3;
            int ih2 = oh + kh2 - 1; ih2 = ih2 < 0 ? 0 : (ih2 > 55 ? 55 : ih2);
            int iw2 = ow + kw2 - 1; iw2 = iw2 < 0 ? 0 : (iw2 > 55 ? 55 : iw2);
            __builtin_prefetch(xb + (size_t)((k2 & 127) + bkk) * HWPIX + ih2 * IMG_W + iw2, 0, 0);
        }

        // ---- fragment loads (ISA 7.12.2 layouts) + 8 WMMAs ----
        const int row = lane & 15;
        const int hi  = lane >> 4;
        Frag af[2];
#pragma unroll
        for (int mi = 0; mi < 2; ++mi) {
            const int mb = wave + mi * 8;
            const _Float16* ap = &As[buf][(mb * 16 + row) * 40 + hi * 8];
            af[mi].h[0] = *(const v8h*)ap;
            af[mi].h[1] = *(const v8h*)(ap + 16);
        }
#pragma unroll
        for (int nb = 0; nb < 4; ++nb) {
            Frag bf;
            const _Float16* bp = &Bs[(nb * 16 + row) * 40 + hi * 8];
            bf.h[0] = *(const v8h*)bp;
            bf.h[1] = *(const v8h*)(bp + 16);
            acc[0][nb] = __builtin_amdgcn_wmma_f32_16x16x32_f16(false, af[0].v, false, bf.v,
                                                                (short)0, acc[0][nb], false, false);
            acc[1][nb] = __builtin_amdgcn_wmma_f32_16x16x32_f16(false, af[1].v, false, bf.v,
                                                                (short)0, acc[1][nb], false, false);
        }
        __syncthreads();
    }

    // ---- store with fused bias: C layout VGPR r -> M = r + 8*(lane>>4), N = lane&15
    const int col = lane & 15;
    const int hi  = lane >> 4;
    float* outb = out + (size_t)b * C_OUT * HWPIX;
#pragma unroll
    for (int mi = 0; mi < 2; ++mi) {
        const int obase = (wave + mi * 8) * 16 + hi * 8;
#pragma unroll
        for (int r = 0; r < 8; ++r) {
            const int o = obase + r;
            const float bb = bias_b[b * C_OUT + o];
#pragma unroll
            for (int nb = 0; nb < 4; ++nb)
                outb[(size_t)o * HWPIX + n0 + nb * 16 + col] = acc[mi][nb][r] + bb;
        }
    }
}

extern "C" void kernel_launch(void* const* d_in, const int* in_sizes, int n_in,
                              void* d_out, int out_size, void* d_ws, size_t ws_size,
                              hipStream_t stream) {
    (void)in_sizes; (void)n_in; (void)out_size; (void)ws_size;
    const float* x      = (const float*)d_in[0];
    const float* weight = (const float*)d_in[1];
    const float* rw     = (const float*)d_in[2];
    const float* rb     = (const float*)d_in[3];
    const float* bias   = (const float*)d_in[4];
    float* out = (float*)d_out;

    float* wsf      = (float*)d_ws;
    float* pooled   = wsf;
    float* routing  = wsf + 4096;
    float* bias_b   = wsf + 4352;
    _Float16* w16   = (_Float16*)(wsf + 12544);

    cc_pool<<<dim3(C_IN, NB), 256, 0, stream>>>(x, pooled);
    cc_route<<<dim3(NB), 256, 0, stream>>>(pooled, rw, rb, bias, routing, bias_b);
    cc_mixw<<<dim3(NPARAMS / 256, NB), 256, 0, stream>>>(weight, routing, w16);
    cc_conv_wmma<<<dim3(HWPIX / 64, NB), 256, 0, stream>>>(x, w16, bias_b, out);
}
